// sparse_bottleneck_68659347194199
// MI455X (gfx1250) — matrix operand order repaired and checked by reading, not compiled, not run on hardware
//
#include <hip/hip_runtime.h>

// ---------------------------------------------------------------------------
// Types for CDNA5 WMMA (wave32): D(f32 16x16) = A(bf16 16x32) x B(bf16 32x16) + C
// ---------------------------------------------------------------------------
typedef __attribute__((ext_vector_type(16))) __bf16 v16bf;
typedef __attribute__((ext_vector_type(8)))  __bf16 v8bf;
typedef __attribute__((ext_vector_type(8)))  float  v8f;

// A fragment (16x32 bf16): lane<16 holds row M=lane, K in {kb+0..7, kb+16..23};
// lane>=16 holds row M=lane-16, K in {kb+8..15, kb+24..31}.
__device__ __forceinline__ v16bf load_a_frag_bf16(const __bf16* __restrict__ row,
                                                  int kb, int half) {
  const __bf16* p0 = row + kb + half * 8;
  const __bf16* p1 = row + kb + 16 + half * 8;
  v8bf lo = *(const v8bf*)p0;
  v8bf hi = *(const v8bf*)p1;
  v16bf a;
#pragma unroll
  for (int i = 0; i < 8; ++i) { a[i] = lo[i]; a[8 + i] = hi[i]; }
  return a;
}

// Same A layout but sourced from fp32 (converted to bf16 on the fly).
__device__ __forceinline__ v16bf load_a_frag_f32(const float* __restrict__ row,
                                                 int kb, int half) {
  const float* p0 = row + kb + half * 8;
  const float* p1 = row + kb + 16 + half * 8;
  v16bf a;
#pragma unroll
  for (int i = 0; i < 8; ++i) {
    a[i]     = (__bf16)p0[i];
    a[8 + i] = (__bf16)p1[i];
  }
  return a;
}

// B fragment (32x16 bf16) from LDS weights stored transposed as wT[col][K]
// (K contiguous, row stride KS). lane<16: col=lane, K in {kb+0..7, kb+16..23};
// lane>=16: col=lane-16, K in {kb+8..15, kb+24..31}.
__device__ __forceinline__ v16bf load_b_frag_lds(const __bf16* __restrict__ wT,
                                                 int col, int KS, int kb, int half) {
  const __bf16* p = wT + (size_t)col * KS + kb + half * 8;
  v16bf b;
#pragma unroll
  for (int i = 0; i < 8; ++i) { b[i] = p[i]; b[8 + i] = p[16 + i]; }
  return b;
}

__device__ __forceinline__ float relu_f(float x) { return x > 0.f ? x : 0.f; }

// ---------------------------------------------------------------------------
// Kernel 1: out1[N,64](bf16) = relu(features[N,256] @ w1[256,64] * s1 + b1)
// Block = 128 threads (4 waves); each wave: 16 rows x 64 cols.
// ---------------------------------------------------------------------------
__global__ void __launch_bounds__(128)
sb_conv1_kernel(const float* __restrict__ feat, const float* __restrict__ w1,
                const float* __restrict__ s1, const float* __restrict__ b1,
                __bf16* __restrict__ out1, int N) {
  __shared__ __bf16 w1T[64 * 256];  // [col][k], 32 KB
  const int t = threadIdx.x;

  // Zero the rulebook sentinel row (index N) once.
  if (blockIdx.x == 0 && t < 64) out1[(size_t)N * 64 + t] = (__bf16)0.f;

  // Cooperative transposed load of w1 (f32 [256][64] -> bf16 [64][256]).
  for (int i = t; i < 256 * 64; i += 128) {
    int k = i >> 6, c = i & 63;
    w1T[c * 256 + k] = (__bf16)w1[i];
  }
  __syncthreads();

  const int lane = t & 31, wave = t >> 5;
  const int half = lane >> 4, m = lane & 15;

  int rowa = blockIdx.x * 64 + wave * 16 + m;
  if (rowa >= N) rowa = N - 1;  // keep EXEC all-ones through WMMA
  const float* arow = feat + (size_t)rowa * 256;

  v8f acc[4] = {};
#pragma unroll
  for (int kb = 0; kb < 256; kb += 32) {
    v16bf a = load_a_frag_f32(arow, kb, half);
    // Load all four B fragments into distinct registers, then issue four
    // independent WMMAs: LDS loads clause together, no per-WMMA dscnt wait.
    v16bf b0 = load_b_frag_lds(w1T, 0 * 16 + m, 256, kb, half);
    v16bf b1 = load_b_frag_lds(w1T, 1 * 16 + m, 256, kb, half);
    v16bf b2 = load_b_frag_lds(w1T, 2 * 16 + m, 256, kb, half);
    v16bf b3 = load_b_frag_lds(w1T, 3 * 16 + m, 256, kb, half);
    acc[0] = __builtin_amdgcn_wmma_f32_16x16x32_bf16(false, a, false, b0,
                                                     (short)0, acc[0], false, false);
    acc[1] = __builtin_amdgcn_wmma_f32_16x16x32_bf16(false, a, false, b1,
                                                     (short)0, acc[1], false, false);
    acc[2] = __builtin_amdgcn_wmma_f32_16x16x32_bf16(false, a, false, b2,
                                                     (short)0, acc[2], false, false);
    acc[3] = __builtin_amdgcn_wmma_f32_16x16x32_bf16(false, a, false, b3,
                                                     (short)0, acc[3], false, false);
  }

  const int rbase = blockIdx.x * 64 + wave * 16 + half * 8;
#pragma unroll
  for (int ct = 0; ct < 4; ++ct) {
    const int c = ct * 16 + m;
    const float sc = s1[c], bi = b1[c];
#pragma unroll
    for (int j = 0; j < 8; ++j) {
      const int r = rbase + j;
      if (r < N) out1[(size_t)r * 64 + c] = (__bf16)relu_f(acc[ct][j] * sc + bi);
    }
  }
}

// ---------------------------------------------------------------------------
// Kernel 2: out2[n] = relu( (sum_k out1[idx[n,k]] @ w2[k]) * s2 + b2 )
// All 9 transposed weight matrices staged once in 72 KB LDS (gfx1250 allows
// up to 320 KB per workgroup); single barrier, then the 9x(gather + 8 WMMA)
// chain pipelines with no further synchronization.
// ---------------------------------------------------------------------------
__global__ void __launch_bounds__(128)
sb_conv2_kernel(const __bf16* __restrict__ out1, const int* __restrict__ nidx,
                const float* __restrict__ w2, const float* __restrict__ s2,
                const float* __restrict__ b2, __bf16* __restrict__ out2, int N) {
  __shared__ __bf16 w2T[9 * 64 * 64];  // [k][d][c], 72 KB
  const int t = threadIdx.x;

  for (int i = t; i < 9 * 64 * 64; i += 128) {
    int k = i >> 12, r = i & 4095;
    int c = r >> 6, d = r & 63;
    w2T[k * 4096 + d * 64 + c] = (__bf16)w2[i];
  }
  __syncthreads();

  const int lane = t & 31, wave = t >> 5;
  const int half = lane >> 4, m = lane & 15;

  int rowa = blockIdx.x * 64 + wave * 16 + m;
  if (rowa >= N) rowa = N - 1;

  // Preload the 9 rulebook indices for this lane's row.
  int gk[9];
#pragma unroll
  for (int k = 0; k < 9; ++k) gk[k] = nidx[(size_t)rowa * 9 + k];

  v8f acc[4] = {};
#pragma unroll
  for (int k = 0; k < 9; ++k) {
    const __bf16* arow = out1 + (size_t)gk[k] * 64;  // L2-resident gather
    const __bf16* wk = w2T + k * 4096;
#pragma unroll
    for (int kk = 0; kk < 64; kk += 32) {
      v16bf a  = load_a_frag_bf16(arow, kk, half);
      v16bf b0 = load_b_frag_lds(wk, 0 * 16 + m, 64, kk, half);
      v16bf b1 = load_b_frag_lds(wk, 1 * 16 + m, 64, kk, half);
      v16bf b2 = load_b_frag_lds(wk, 2 * 16 + m, 64, kk, half);
      v16bf b3 = load_b_frag_lds(wk, 3 * 16 + m, 64, kk, half);
      acc[0] = __builtin_amdgcn_wmma_f32_16x16x32_bf16(false, a, false, b0,
                                                       (short)0, acc[0], false, false);
      acc[1] = __builtin_amdgcn_wmma_f32_16x16x32_bf16(false, a, false, b1,
                                                       (short)0, acc[1], false, false);
      acc[2] = __builtin_amdgcn_wmma_f32_16x16x32_bf16(false, a, false, b2,
                                                       (short)0, acc[2], false, false);
      acc[3] = __builtin_amdgcn_wmma_f32_16x16x32_bf16(false, a, false, b3,
                                                       (short)0, acc[3], false, false);
    }
  }

  const int rbase = blockIdx.x * 64 + wave * 16 + half * 8;
#pragma unroll
  for (int ct = 0; ct < 4; ++ct) {
    const int c = ct * 16 + m;
    const float sc = s2[c], bi = b2[c];
#pragma unroll
    for (int j = 0; j < 8; ++j) {
      const int r = rbase + j;
      if (r < N) out2[(size_t)r * 64 + c] = (__bf16)relu_f(acc[ct][j] * sc + bi);
    }
  }
}

// ---------------------------------------------------------------------------
// Kernel 3: out[N,256] = relu(out2[N,64] @ w3[64,256] * s3 + b3 + features)
// Block = 4 waves x 16 rows x all 256 cols (16 col-tiles, A kept in regs).
// ---------------------------------------------------------------------------
__global__ void __launch_bounds__(128)
sb_conv3_kernel(const __bf16* __restrict__ out2, const float* __restrict__ w3,
                const float* __restrict__ s3, const float* __restrict__ b3,
                const float* __restrict__ feat, float* __restrict__ out, int N) {
  __shared__ __bf16 w3T[256 * 64];  // [d][k], 32 KB
  const int t = threadIdx.x;
  for (int i = t; i < 64 * 256; i += 128) {
    int k = i >> 8, d = i & 255;
    w3T[d * 64 + k] = (__bf16)w3[i];
  }
  __syncthreads();

  const int lane = t & 31, wave = t >> 5;
  const int half = lane >> 4, m = lane & 15;

  int rowa = blockIdx.x * 64 + wave * 16 + m;
  if (rowa >= N) rowa = N - 1;
  const __bf16* arow = out2 + (size_t)rowa * 64;

  v16bf a0 = load_a_frag_bf16(arow, 0, half);
  v16bf a1 = load_a_frag_bf16(arow, 32, half);

  const int rbase = blockIdx.x * 64 + wave * 16 + half * 8;
#pragma unroll 4
  for (int ct = 0; ct < 16; ++ct) {
    // Hoist both B fragments; the two WMMAs chain through the accumulator
    // (D->C forwarding, no D->A/B hazard spacing needed).
    v16bf bb0 = load_b_frag_lds(w3T, ct * 16 + m, 64, 0, half);
    v16bf bb1 = load_b_frag_lds(w3T, ct * 16 + m, 64, 32, half);
    v8f acc = {};
    acc = __builtin_amdgcn_wmma_f32_16x16x32_bf16(false, a0, false, bb0,
                                                  (short)0, acc, false, false);
    acc = __builtin_amdgcn_wmma_f32_16x16x32_bf16(false, a1, false, bb1,
                                                  (short)0, acc, false, false);

    const int c = ct * 16 + m;
    const float sc = s3[c], bi = b3[c];
#pragma unroll
    for (int j = 0; j < 8; ++j) {
      const int r = rbase + j;
      if (r < N) {
        float v = acc[j] * sc + bi + feat[(size_t)r * 256 + c];
        out[(size_t)r * 256 + c] = relu_f(v);
      }
    }
  }
}

// ---------------------------------------------------------------------------
// Launch: conv1 -> conv2 (rulebook gather) -> conv3, all on `stream`.
// ws layout: out1 bf16 [(N+1) x 64] | out2 bf16 [N x 64]  (~25.6 MB total)
// ---------------------------------------------------------------------------
extern "C" void kernel_launch(void* const* d_in, const int* in_sizes, int n_in,
                              void* d_out, int out_size, void* d_ws, size_t ws_size,
                              hipStream_t stream) {
  const float* feat = (const float*)d_in[0];
  const int*   nidx = (const int*)d_in[1];
  const float* w1 = (const float*)d_in[2];
  const float* s1 = (const float*)d_in[3];
  const float* b1 = (const float*)d_in[4];
  const float* w2 = (const float*)d_in[5];
  const float* s2 = (const float*)d_in[6];
  const float* b2 = (const float*)d_in[7];
  const float* w3 = (const float*)d_in[8];
  const float* s3 = (const float*)d_in[9];
  const float* b3 = (const float*)d_in[10];
  float* out = (float*)d_out;

  const int N = in_sizes[0] / 256;

  __bf16* o1 = (__bf16*)d_ws;
  size_t o1_bytes = (((size_t)(N + 1) * 64 * 2) + 255) & ~(size_t)255;
  __bf16* o2 = (__bf16*)((char*)d_ws + o1_bytes);

  const int blocks = (N + 63) / 64;
  sb_conv1_kernel<<<blocks, 128, 0, stream>>>(feat, w1, s1, b1, o1, N);
  sb_conv2_kernel<<<blocks, 128, 0, stream>>>(o1, nidx, w2, s2, b2, o2, N);
  sb_conv3_kernel<<<blocks, 128, 0, stream>>>(o2, w3, s3, b3, feat, out, N);
}
